// PointCNN_20366734917771
// MI455X (gfx1250) — matrix operand order repaired and checked by reading, not compiled, not run on hardware
//
#include <hip/hip_runtime.h>
#include <hip/hip_bf16.h>

typedef _Float16 v8h  __attribute__((ext_vector_type(8)));
typedef _Float16 v16h __attribute__((ext_vector_type(16)));
typedef float    v8f  __attribute__((ext_vector_type(8)));

#define BN_S 0.9999950000374997f   // 1/sqrt(1+1e-5)
#define KCAP 48

__device__ __forceinline__ float eluf(float x) {
    return x > 0.f ? x : (__expf(x) - 1.f);
}

// build a 16-half fragment from two contiguous 8-half (16B) chunks
__device__ __forceinline__ v16h ld2(const _Float16* __restrict__ p0,
                                    const _Float16* __restrict__ p1) {
    v8h lo = *(const v8h*)p0;
    v8h hi = *(const v8h*)p1;
    v16h r;
#pragma unroll
    for (int i = 0; i < 8; ++i) { r[i] = lo[i]; r[i + 8] = hi[i]; }
    return r;
}

// ---------------------------------------------------------------- sampling
__global__ void sample_kernel(const float* __restrict__ p, float* __restrict__ q,
                              int B, int N, int M) {
    int t = blockIdx.x * blockDim.x + threadIdx.x;
    if (t >= B * M * 3) return;
    int d = t % 3, m = (t / 3) % M, b = t / (3 * M);
    int n = (int)(((long long)m * N) / M);
    q[((size_t)b * M + m) * 3 + d] = p[((size_t)b * N + n) * 3 + d];
}

// ---------------------------------------------------------------- dilated kNN
__global__ void knn_kernel(const float* __restrict__ p, const float* __restrict__ q,
                           int* __restrict__ idxOut, int B, int N, int M, int k, int dil) {
    int t = blockIdx.x * blockDim.x + threadIdx.x;
    if (t >= B * M) return;
    int b = t / M, m = t % M;
    const float* qp = q + ((size_t)b * M + m) * 3;
    float qx = qp[0], qy = qp[1], qz = qp[2];
    int K = k * dil;
    float bd[KCAP]; int bi[KCAP];
    for (int i = 0; i < KCAP; ++i) { bd[i] = 3.4e38f; bi[i] = 0; }
    const float* pb = p + (size_t)b * N * 3;
    for (int n = 0; n < N; ++n) {
        float dx = pb[n * 3 + 0] - qx;
        float dy = pb[n * 3 + 1] - qy;
        float dz = pb[n * 3 + 2] - qz;
        float d2 = dx * dx + dy * dy + dz * dz;
        if (d2 < bd[K - 1]) {
            int j = K - 1;
            while (j > 0 && bd[j - 1] > d2) { bd[j] = bd[j - 1]; bi[j] = bi[j - 1]; --j; }
            bd[j] = d2; bi[j] = n;
        }
    }
    int* o = idxOut + ((size_t)b * M + m) * k;
    for (int j = 0; j < k; ++j) o[j] = bi[j * dil];
}

// ---------------------------------------------------------------- rel + mlp1 layer 1
__global__ void lift_a(const float* __restrict__ p, const float* __restrict__ q,
                       const int* __restrict__ idx,
                       const float* __restrict__ wd1, const float* __restrict__ g1,
                       const float* __restrict__ b1,
                       float* __restrict__ rel, float* __restrict__ h1,
                       int B, int N, int M, int k, int cmid) {
    int t = blockIdx.x * blockDim.x + threadIdx.x;
    if (t >= B * M * k) return;
    int b = t / (M * k), m = (t / k) % M;
    int n = idx[t];
    const float* pp = p + ((size_t)b * N + n) * 3;
    const float* qp = q + ((size_t)b * M + m) * 3;
    float rx = pp[0] - qp[0], ry = pp[1] - qp[1], rz = pp[2] - qp[2];
    rel[(size_t)t * 3 + 0] = rx;
    rel[(size_t)t * 3 + 1] = ry;
    rel[(size_t)t * 3 + 2] = rz;
    float* ho = h1 + (size_t)t * cmid;
    for (int c = 0; c < cmid; ++c) {
        float v = rx * wd1[c] + ry * wd1[cmid + c] + rz * wd1[2 * cmid + c];
        v = v * (g1[c] * BN_S) + b1[c];
        ho[c] = eluf(v);
    }
}

// ---------------------------------------------------------------- mlp1 layer 2 + x gather -> f
__global__ void lift_b(const float* __restrict__ h1, const float* __restrict__ x,
                       const int* __restrict__ idx,
                       const float* __restrict__ wd2, const float* __restrict__ g2,
                       const float* __restrict__ b2,
                       float* __restrict__ f, int B, int N, int M, int k, int cmid, int cin) {
    int t = blockIdx.x * blockDim.x + threadIdx.x;
    if (t >= B * M * k) return;
    int b = t / (M * k);
    int C = cmid + cin;
    const float* h = h1 + (size_t)t * cmid;
    float* fo = f + (size_t)t * C;
    for (int c = 0; c < cmid; ++c) {
        float acc = 0.f;
        for (int i = 0; i < cmid; ++i) acc += h[i] * wd2[i * cmid + c];
        acc = acc * (g2[c] * BN_S) + b2[c];
        fo[c] = eluf(acc);
    }
    int n = idx[t];
    for (int ci = 0; ci < cin; ++ci)
        fo[cmid + ci] = x[((size_t)b * cin + ci) * N + n];
}

// ---------------------------------------------------------------- zero f16 buffer
__global__ void zero_h(_Float16* __restrict__ p, int n) {
    int t = blockIdx.x * blockDim.x + threadIdx.x;
    if (t < n) p[t] = (_Float16)0.f;
}

// ---------------------------------------------------------------- X-transform chain + fx = T @ f
__global__ __launch_bounds__(128) void xform_fx(
    const float* __restrict__ rel, const float* __restrict__ f,
    const float* __restrict__ wt1, const float* __restrict__ gt1, const float* __restrict__ bt1,
    const float* __restrict__ wt2, const float* __restrict__ gt2, const float* __restrict__ bt2,
    const float* __restrict__ wt3, const float* __restrict__ gt3, const float* __restrict__ bt3,
    _Float16* __restrict__ Afx, int B, int M, int k, int C, int kpad) {
    __shared__ float rs[4][64];
    __shared__ float t1s[4][256];
    __shared__ float t2s[4][256];
    int lp = threadIdx.x >> 5;
    int lane = threadIdx.x & 31;
    int pt = blockIdx.x * 4 + lp;
    bool act = pt < B * M;
    int k3 = k * 3, kk2 = k * k;
    if (act) {
        const float* rp = rel + (size_t)pt * k3;
        for (int e = lane; e < k3; e += 32) rs[lp][e] = rp[e];
    }
    __syncthreads();
    if (act) {
        for (int e = lane; e < kk2; e += 32) {
            float acc = 0.f;
            for (int d = 0; d < k3; ++d) acc += rs[lp][d] * wt1[d * kk2 + e];
            acc = acc * (gt1[e] * BN_S) + bt1[e];
            t1s[lp][e] = eluf(acc);
        }
    }
    __syncthreads();
    if (act) {
        for (int e = lane; e < kk2; e += 32) {
            int g = e / k, jj = e % k;
            float acc = 0.f;
            for (int i = 0; i < k; ++i) acc += t1s[lp][g * k + i] * wt2[(g * k + i) * k + jj];
            acc = acc * (gt2[e] * BN_S) + bt2[e];
            t2s[lp][e] = eluf(acc);
        }
    }
    __syncthreads();
    if (act) {
        for (int e = lane; e < kk2; e += 32) {
            int g = e / k, jj = e % k;
            float acc = 0.f;
            for (int i = 0; i < k; ++i) acc += t2s[lp][g * k + i] * wt3[(g * k + i) * k + jj];
            t1s[lp][e] = acc * (gt3[e] * BN_S) + bt3[e];   // T (no activation)
        }
    }
    __syncthreads();
    if (act) {
        const float* fp = f + (size_t)pt * k * C;
        _Float16* ao = Afx + (size_t)pt * kpad;
        int kC = k * C;
        for (int e = lane; e < kC; e += 32) {
            int i = e / C, c = e % C;
            float acc = 0.f;
            for (int j = 0; j < k; ++j) acc += t1s[lp][i * k + j] * fp[j * C + c];
            ao[i * C + c] = (_Float16)acc;
        }
    }
}

// -------------------------------------------------- conv weight repack: [Cout,C,k] -> f16 [colsPad][kpad] (K = i*C+c)
__global__ void repack_conv(const float* __restrict__ w, _Float16* __restrict__ Wt,
                            int cout, int colsPad, int C, int k, int kpad) {
    int t = blockIdx.x * blockDim.x + threadIdx.x;
    if (t >= colsPad * kpad) return;
    int o = t / kpad, kidx = t % kpad;
    float v = 0.f;
    if (o < cout && kidx < k * C) {
        int i = kidx / C, c = kidx % C;
        v = w[((size_t)o * C + c) * k + i];
    }
    Wt[(size_t)o * kpad + kidx] = (_Float16)v;
}

// -------------------------------------------------- mat weight repack: [K][N] f32 -> f16 [Npad][K]
__global__ void repack_mat(const float* __restrict__ w, _Float16* __restrict__ Wt,
                           int K, int Ncols, int Npad) {
    int t = blockIdx.x * blockDim.x + threadIdx.x;
    if (t >= Npad * K) return;
    int n = t / K, kidx = t % K;
    float v = (n < Ncols) ? w[(size_t)kidx * Ncols + n] : 0.f;
    Wt[(size_t)n * K + kidx] = (_Float16)v;
}

// ---------------------------------------------------------------- WMMA f16 GEMM: D = act(bn(A @ Bt^T))
// A [rows][lda] f16 row-major, Bt [colsPad][ldb] f16 (row = output col, contiguous K).
// One wave computes a 16x32 output (two col tiles, shared A fragment); colsPad is a
// multiple of 32 so both WMMAs are unconditional (EXEC stays all-1, per ISA 7.12).
// Register double-buffering with a peeled final K-step keeps loads ahead of the
// matrix pipe. Fragment layouts per CDNA5 ISA 7.12.2.
__global__ __launch_bounds__(256) void gemm_wmma(
    const _Float16* __restrict__ A, int lda,
    const _Float16* __restrict__ Bt, int ldb,
    float* __restrict__ outF32, int Mper,          // batched-transposed store [b][col][m]
    _Float16* __restrict__ outF16, int ldo,        // row-major f16 store
    const float* __restrict__ gamma, const float* __restrict__ beta,
    int actmode, int rows, int cols, int colsPad, int kdim) {
    int wid = blockIdx.x * 8 + (threadIdx.x >> 5);
    int lane = threadIdx.x & 31;
    int tilesC2 = colsPad >> 5;
    int tr  = wid / tilesC2;
    int tc0 = (wid % tilesC2) * 2;
    if (tr * 16 >= rows) return;                    // wave-uniform: EXEC stays all-1 for WMMA

    int arow  = tr * 16 + (lane & 15);
    int abase = (lane < 16) ? 0 : 8;                // A: lanes 0-15 K={0..7,16..23}, 16-31 K={8..15,24..31}
    int brow  = tc0 * 16 + (lane & 15);
    int bbase = (lane < 16) ? 0 : 8;                // B: lanes 0-15 K={0..7,16..23}, 16-31 K={8..15,24..31}
    const _Float16* Ap  = A  + (size_t)arow * lda + abase;
    const _Float16* Bp0 = Bt + (size_t)brow * ldb + bbase;
    const _Float16* Bp1 = Bp0 + (size_t)16 * ldb;

    // prime the pipeline
    v16h a  = ld2(Ap, Ap + 16);
    v16h b0 = ld2(Bp0, Bp0 + 16);
    v16h b1 = ld2(Bp1, Bp1 + 16);

    v8f c0 = {}, c1 = {};
    int kk = 0;
#pragma unroll 2
    for (; kk + 32 < kdim; kk += 32) {              // all loads unconditional; last step peeled
        v16h an  = ld2(Ap + kk + 32, Ap + kk + 48);
        v16h b0n = ld2(Bp0 + kk + 32, Bp0 + kk + 48);
        v16h b1n = ld2(Bp1 + kk + 32, Bp1 + kk + 48);
        if (kk + 96 <= kdim) {                      // global_prefetch_b8, near-cache
            __builtin_prefetch(Ap + kk + 64, 0, 3);
            __builtin_prefetch(Bp0 + kk + 64, 0, 3);
        }
        c0 = __builtin_amdgcn_wmma_f32_16x16x32_f16(false, a, false, b0,
                                                    (short)0, c0, false, false);
        c1 = __builtin_amdgcn_wmma_f32_16x16x32_f16(false, a, false, b1,
                                                    (short)0, c1, false, false);
        a = an; b0 = b0n; b1 = b1n;
    }
    c0 = __builtin_amdgcn_wmma_f32_16x16x32_f16(false, a, false, b0,
                                                (short)0, c0, false, false);
    c1 = __builtin_amdgcn_wmma_f32_16x16x32_f16(false, a, false, b1,
                                                (short)0, c1, false, false);

#pragma unroll
    for (int half = 0; half < 2; ++half) {
        int n = (tc0 + half) * 16 + (lane & 15);
        const v8f& cc = half ? c1 : c0;
        if (n < cols) {                             // per-lane store guard (no WMMA inside)
#pragma unroll
            for (int r = 0; r < 8; ++r) {
                int m = tr * 16 + r + ((lane < 16) ? 0 : 8);
                float v = cc[r];
                if (gamma) v = v * (gamma[n] * BN_S) + beta[n];
                if (actmode == 2) v = v > 0.f ? v : 0.f;    // relu
                if (outF32) {
                    int bb = m / Mper, ml = m % Mper;
                    outF32[((size_t)bb * cols + n) * Mper + ml] = v;
                }
                if (outF16) outF16[(size_t)m * ldo + n] = (_Float16)v;
            }
        }
    }
}

// ---------------------------------------------------------------- final fc + mean pool
__global__ void fc_mean(const _Float16* __restrict__ h, const float* __restrict__ fw,
                        const float* __restrict__ fb, float* __restrict__ out,
                        int B, int M, int K, int ncls) {
    int t = blockIdx.x * blockDim.x + threadIdx.x;
    if (t >= B * ncls) return;
    int b = t / ncls, cls = t % ncls;
    float acc = 0.f;
    for (int m = 0; m < M; ++m) {
        const _Float16* hp = h + ((size_t)b * M + m) * K;
        float d = 0.f;
        for (int c = 0; c < K; ++c) d += (float)hp[c] * fw[c * ncls + cls];
        acc += d;
    }
    out[t] = acc / (float)M + fb[cls];
}

// ================================================================ host
extern "C" void kernel_launch(void* const* d_in, const int* in_sizes, int n_in,
                              void* d_out, int out_size, void* d_ws, size_t ws_size,
                              hipStream_t stream) {
    (void)in_sizes; (void)n_in; (void)out_size; (void)ws_size;
    const float* P  = (const float*)d_in[0];
    const float* X0 = (const float*)d_in[1];

    char* wsp = (char*)d_ws;
    size_t off = 0;
    auto carve = [&](size_t bytes) -> void* {
        void* r = wsp + off;
        off += (bytes + 255) & ~(size_t)255;
        return r;
    };
    float*    q1   = (float*)carve(8UL * 1024 * 3 * 4);
    float*    q2   = (float*)carve(8UL * 384 * 3 * 4);
    float*    q3   = (float*)carve(8UL * 128 * 3 * 4);
    int*      idxB = (int*)carve(8UL * 1024 * 8 * 4);
    float*    relB = (float*)carve(8UL * 1024 * 8 * 3 * 4);
    float*    h1B  = (float*)carve(8UL * 128 * 16 * 96 * 4);
    float*    fB   = (float*)carve(8UL * 128 * 16 * 288 * 4);
    _Float16* Afx  = (_Float16*)carve(1024UL * 4608 * 2);
    _Float16* Wt   = (_Float16*)carve(384UL * 4608 * 2);
    float*    xA   = (float*)carve(8UL * 48 * 1024 * 4);
    float*    xB   = (float*)carve(8UL * 96 * 384 * 4);
    _Float16* H0   = (_Float16*)carve(1024UL * 384 * 2);
    _Float16* H1   = (_Float16*)carve(1024UL * 256 * 2);
    _Float16* H2   = (_Float16*)carve(1024UL * 128 * 2);

    const int B = 8;
    const int Ns[4]    = {8192, 1024, 384, 128};
    const int Ms[4]    = {1024, 384, 128, 128};
    const int ks[4]    = {8, 12, 16, 16};
    const int dils[4]  = {1, 2, 2, 3};
    const int cins[4]  = {3, 48, 96, 192};
    const int couts[4] = {48, 96, 192, 384};
    const int base[4]  = {2, 18, 34, 50};
    const float* ps[4]    = {P, q1, q2, q3};
    float*       qs[4]    = {q1, q2, q3, q3};
    const float* xs[4]    = {X0, xA, xB, xA};
    float*       xouts[4] = {xA, xB, xA, nullptr};

    for (int s = 0; s < 4; ++s) {
        int N = Ns[s], M = Ms[s], k = ks[s], dil = dils[s];
        int cin = cins[s], cout = couts[s];
        int cmid = cout / 4, C = cmid + cin;
        int Kdim = k * C, Kpad = (Kdim + 31) & ~31;
        int colsPad = (cout + 31) & ~31;
        int bi = base[s];
        const float* pS = ps[s];
        float* qS = qs[s];

        if (s < 3)
            sample_kernel<<<(B * M * 3 + 255) / 256, 256, 0, stream>>>(pS, qS, B, N, M);
        knn_kernel<<<(B * M + 255) / 256, 256, 0, stream>>>(pS, qS, idxB, B, N, M, k, dil);
        lift_a<<<(B * M * k + 255) / 256, 256, 0, stream>>>(
            pS, qS, idxB,
            (const float*)d_in[bi + 0], (const float*)d_in[bi + 1], (const float*)d_in[bi + 2],
            relB, h1B, B, N, M, k, cmid);
        lift_b<<<(B * M * k + 255) / 256, 256, 0, stream>>>(
            h1B, xs[s], idxB,
            (const float*)d_in[bi + 3], (const float*)d_in[bi + 4], (const float*)d_in[bi + 5],
            fB, B, N, M, k, cmid, cin);
        if (Kpad != Kdim)
            zero_h<<<(B * M * Kpad + 255) / 256, 256, 0, stream>>>(Afx, B * M * Kpad);
        xform_fx<<<(B * M + 3) / 4, 128, 0, stream>>>(
            relB, fB,
            (const float*)d_in[bi + 6], (const float*)d_in[bi + 7], (const float*)d_in[bi + 8],
            (const float*)d_in[bi + 9], (const float*)d_in[bi + 10], (const float*)d_in[bi + 11],
            (const float*)d_in[bi + 12], (const float*)d_in[bi + 13], (const float*)d_in[bi + 14],
            Afx, B, M, k, C, Kpad);
        repack_conv<<<(colsPad * Kpad + 255) / 256, 256, 0, stream>>>(
            (const float*)d_in[bi + 15], Wt, cout, colsPad, C, k, Kpad);

        int rows = B * M;
        int tiles = (rows / 16) * (colsPad / 32);
        gemm_wmma<<<(tiles + 7) / 8, 256, 0, stream>>>(
            Afx, Kpad, Wt, Kpad,
            xouts[s], M,
            (s == 3) ? H0 : (_Float16*)nullptr, 384,
            (const float*)nullptr, (const float*)nullptr, 0,
            rows, cout, colsPad, Kpad);
    }

    // classifier MLP: 384 -> 256 -> 128 (bn+relu via WMMA), then fc 128 -> 40 + mean
    repack_mat<<<(256 * 384 + 255) / 256, 256, 0, stream>>>(
        (const float*)d_in[66], Wt, 384, 256, 256);
    {
        int tiles = (1024 / 16) * (256 / 32);
        gemm_wmma<<<(tiles + 7) / 8, 256, 0, stream>>>(
            H0, 384, Wt, 384,
            (float*)nullptr, 1, H1, 256,
            (const float*)d_in[67], (const float*)d_in[68], 2,
            1024, 256, 256, 384);
    }
    repack_mat<<<(128 * 256 + 255) / 256, 256, 0, stream>>>(
        (const float*)d_in[69], Wt, 256, 128, 128);
    {
        int tiles = (1024 / 16) * (128 / 32);
        gemm_wmma<<<(tiles + 7) / 8, 256, 0, stream>>>(
            H1, 256, Wt, 256,
            (float*)nullptr, 1, H2, 128,
            (const float*)d_in[70], (const float*)d_in[71], 2,
            1024, 128, 128, 256);
    }
    fc_mean<<<(8 * 40 + 255) / 256, 256, 0, stream>>>(
        H2, (const float*)d_in[72], (const float*)d_in[73], (float*)d_out,
        8, 128, 128, 40);
}
